// AttentionBlock_67035849556061
// MI455X (gfx1250) — matrix operand order repaired and checked by reading, not compiled, not run on hardware
//
#include <hip/hip_runtime.h>
#include <hip/hip_bf16.h>
#include <math.h>

// ---------------------------------------------------------------------------
// CDNA5 (gfx1250) WMMA types
// ---------------------------------------------------------------------------
typedef __attribute__((ext_vector_type(16))) __bf16 v16bf;
typedef __attribute__((ext_vector_type(8)))  float  v8f;

#define C_DIM  640
#define NHEADS 8
#define DH     80
#define DP     96          // head dim padded to multiple of 32 for WMMA K
#define HWPX   1024        // 32*32 pixels per image
#define NB     8
#define MTOK   (NB*HWPX)   // 8192 tokens

#define KB     64          // K-chunk staged in LDS per pipeline stage
#define LROW   72          // padded LDS row stride in bf16 (144B, kills bank conflicts)

static __device__ __forceinline__ v8f wmma_bf16(v16bf a, v16bf b, v8f c) {
  // D = A(16x32 bf16) * B(32x16 bf16) + C(16x16 f32)
  return __builtin_amdgcn_wmma_f32_16x16x32_bf16(false, a, false, b, (short)0, c,
                                                 false, false);
}

// A-matrix 16x32 bf16 fragment: lanes 0-15 row M=lane, K=[0..7]∪[16..23];
// lanes 16-31 same rows, K=[8..15]∪[24..31].  (ISA 7.12.2)
static __device__ __forceinline__ v16bf ldfragA(const __bf16* row, int k, int hs) {
  union { uint4 q[2]; v16bf v; } u;
  u.q[0] = *(const uint4*)(row + k + hs * 8);
  u.q[1] = *(const uint4*)(row + k + 16 + hs * 8);
  return u.v;
}
// B-matrix 32x16 bf16 fragment: lane holds column N=lane%16,
// K=[0..7]∪[16..23] for lanes 0-15, K=[8..15]∪[24..31] for lanes 16-31.
static __device__ __forceinline__ v16bf ldfragB(const __bf16* row, int k, int hs) {
  union { uint4 q[2]; v16bf v; } u;
  const __bf16* p = row + k + hs * 8;
  u.q[0] = *(const uint4*)(p);
  u.q[1] = *(const uint4*)(p + 16);
  return u.v;
}

// ---------------------------------------------------------------------------
// CDNA5 async global->LDS copy (ASYNCcnt-tracked DMA path, ISA §10.7/§15.18.3)
//   vdst = per-lane LDS byte offset, vaddr = per-lane 64-bit global address
// ---------------------------------------------------------------------------
static __device__ __forceinline__ void async_cp16(unsigned lds_off, const void* g) {
  asm volatile("global_load_async_to_lds_b128 %0, %1, off"
               :: "v"(lds_off), "v"(g) : "memory");
}
#define ASYNC_WAIT(n) asm volatile("s_wait_asynccnt " #n ::: "memory")

static __device__ __forceinline__ unsigned lds_byte_off(const void* p) {
  // generic pointer to LDS: low 32 bits are the LDS byte address (ISA §10.2)
  return (unsigned)(unsigned long long)p;
}

// ---------------------------------------------------------------------------
// GroupNorm: x[8,640,32,32] -> token-major bf16 [8192,640], scale+bias applied
// ---------------------------------------------------------------------------
__global__ void gn_kernel(const float* __restrict__ x, const float* __restrict__ s,
                          const float* __restrict__ b, __bf16* __restrict__ out) {
  int n = blockIdx.x >> 5;           // batch
  int g = blockIdx.x & 31;           // group
  int c0 = g * 20;                   // 640/32 = 20 channels per group
  const float* base = x + ((size_t)n * C_DIM + c0) * HWPX;  // contiguous 20*1024
  __shared__ float r1[256], r2[256];
  float s1 = 0.f, s2 = 0.f;
  for (int i = threadIdx.x; i < 20 * HWPX; i += 256) {
    float v = base[i];
    s1 += v; s2 += v * v;
  }
  r1[threadIdx.x] = s1; r2[threadIdx.x] = s2;
  __syncthreads();
  for (int st = 128; st > 0; st >>= 1) {
    if (threadIdx.x < st) { r1[threadIdx.x] += r1[threadIdx.x + st];
                            r2[threadIdx.x] += r2[threadIdx.x + st]; }
    __syncthreads();
  }
  float mean = r1[0] * (1.0f / (20.0f * HWPX));
  float var  = r2[0] * (1.0f / (20.0f * HWPX)) - mean * mean;
  float rinv = rsqrtf(var + 1e-6f);
  for (int i = threadIdx.x; i < 20 * HWPX; i += 256) {
    int ci = i / HWPX, p = i - ci * HWPX;
    int c = c0 + ci;
    float v = (base[i] - mean) * rinv * s[c] + b[c];
    out[((size_t)n * HWPX + p) * C_DIM + c] = (__bf16)v;
  }
}

// ---------------------------------------------------------------------------
// LayerNorm over last dim (640): fp32 in -> bf16 out.  One wave per row.
// ---------------------------------------------------------------------------
__global__ void ln_kernel(const float* __restrict__ x, const float* __restrict__ s,
                          const float* __restrict__ b, __bf16* __restrict__ out, int M) {
  int row  = blockIdx.x * 8 + (threadIdx.x >> 5);
  int lane = threadIdx.x & 31;
  if (row >= M) return;
  const float* xr = x + (size_t)row * C_DIM;
  float vals[20];
  float s1 = 0.f, s2 = 0.f;
  #pragma unroll
  for (int j = 0; j < 20; ++j) {
    float v = xr[lane + 32 * j];
    vals[j] = v; s1 += v; s2 += v * v;
  }
  #pragma unroll
  for (int m = 16; m >= 1; m >>= 1) {
    s1 += __shfl_xor(s1, m, 32);
    s2 += __shfl_xor(s2, m, 32);
  }
  float mean = s1 * (1.0f / 640.0f);
  float var  = s2 * (1.0f / 640.0f) - mean * mean;
  float r    = rsqrtf(var + 1e-5f);
  __bf16* o = out + (size_t)row * C_DIM;
  #pragma unroll
  for (int j = 0; j < 20; ++j) {
    int c = lane + 32 * j;
    o[c] = (__bf16)((vals[j] - mean) * r * s[c] + b[c]);
  }
}

// ---------------------------------------------------------------------------
// Weight prep: transpose [K,N] fp32 -> [N,K] bf16, and straight cast
// ---------------------------------------------------------------------------
__global__ void wt_kernel(const float* __restrict__ w, __bf16* __restrict__ o,
                          int rows_in, int cols_in) {
  size_t total = (size_t)rows_in * cols_in;
  for (size_t i = (size_t)blockIdx.x * 256 + threadIdx.x; i < total;
       i += (size_t)gridDim.x * 256) {
    size_t j = i / rows_in;          // output row (= input col)
    size_t c = i % rows_in;          // output col (= input row)
    o[i] = (__bf16)w[c * cols_in + j];
  }
}
__global__ void cast_kernel(const float* __restrict__ w, __bf16* __restrict__ o,
                            size_t n) {
  for (size_t i = (size_t)blockIdx.x * 256 + threadIdx.x; i < n;
       i += (size_t)gridDim.x * 256) o[i] = (__bf16)w[i];
}

// ---------------------------------------------------------------------------
// Repack kernels for attention operand layouts
// ---------------------------------------------------------------------------
__global__ void repack_qk_kernel(const float* __restrict__ src, __bf16* __restrict__ dst,
                                 int T, int ld, int coff) {
  size_t total = (size_t)NB * NHEADS * T * DP;
  for (size_t i = (size_t)blockIdx.x * 256 + threadIdx.x; i < total;
       i += (size_t)gridDim.x * 256) {
    int d = (int)(i % DP); size_t r = i / DP;
    int t = (int)(r % T);  size_t bh = r / T;
    int h = (int)(bh % NHEADS); int b = (int)(bh / NHEADS);
    float v = (d < DH) ? src[((size_t)b * T + t) * ld + coff + h * DH + d] : 0.f;
    dst[i] = (__bf16)v;
  }
}
__global__ void repack_vt_kernel(const float* __restrict__ src, __bf16* __restrict__ dst,
                                 int T, int Tpad, int ld, int coff) {
  size_t total = (size_t)NB * NHEADS * DH * Tpad;
  for (size_t i = (size_t)blockIdx.x * 256 + threadIdx.x; i < total;
       i += (size_t)gridDim.x * 256) {
    int t = (int)(i % Tpad); size_t r = i / Tpad;
    int d = (int)(r % DH);   size_t bh = r / DH;
    int h = (int)(bh % NHEADS); int b = (int)(bh / NHEADS);
    float v = (t < T) ? src[((size_t)b * T + t) * ld + coff + h * DH + d] : 0.f;
    dst[i] = (__bf16)v;
  }
}
__global__ void ctxpad_kernel(const float* __restrict__ ctx, __bf16* __restrict__ o) {
  size_t total = (size_t)NB * 80 * 512;
  for (size_t i = (size_t)blockIdx.x * 256 + threadIdx.x; i < total;
       i += (size_t)gridDim.x * 256) {
    int c = (int)(i % 512); size_t r = i / 512;
    int t = (int)(r % 80);  int b = (int)(r / 80);
    float v = (t < 77) ? ctx[((size_t)b * 77 + t) * 512 + c] : 0.f;
    o[i] = (__bf16)v;
  }
}

// ---------------------------------------------------------------------------
// NT GEMM with double-buffered async-to-LDS staging.
//   out[M,N] = A[M,K](bf16) * B[N,K]^T(bf16) (+bias)(+residual)
// Block = 4 waves = 64x64 output tile.  Per K-chunk (64): each thread issues
// 4 async b128 copies for the A strip and 4 for the B strip; compute overlaps
// the next chunk's DMA (ASYNCcnt pipelining).  Fragment loads for each K-step
// are batched ahead of the WMMA burst so the matrix pipe runs back-to-back.
// M, N, K multiples of 64 (shapes: 8192/640 x 640/1920/5120 x 640/512/2560).
// ---------------------------------------------------------------------------
__global__ void gemm_nt(const __bf16* __restrict__ A, const __bf16* __restrict__ Bm,
                        const float* __restrict__ bias, const float* __restrict__ res,
                        float* __restrict__ outF, __bf16* __restrict__ outB,
                        int M, int N, int K, int chw) {
  __shared__ __bf16 Ash[2][64 * LROW];
  __shared__ __bf16 Bsh[2][64 * LROW];
  int tid = threadIdx.x;
  int lane = tid & 31, wid = tid >> 5;
  int m0 = blockIdx.x * 64;
  int n0 = blockIdx.y * 64;
  int r = lane & 15, hs = lane >> 4;

  unsigned ldsA0 = lds_byte_off(&Ash[0][0]);
  unsigned ldsB0 = lds_byte_off(&Bsh[0][0]);

  auto issue = [&](int kb, int buf) {
    unsigned abase = ldsA0 + (unsigned)buf * (64 * LROW * 2);
    unsigned bbase = ldsB0 + (unsigned)buf * (64 * LROW * 2);
    #pragma unroll
    for (int j = 0; j < 4; ++j) {
      int e = j * 128 + tid;           // 512 x 16B chunks, 8 per 64-wide row
      int row = e >> 3, c8 = e & 7;
      unsigned d = (unsigned)(row * LROW + c8 * 8) * 2;
      async_cp16(abase + d, A  + (size_t)(m0 + row) * K + kb + c8 * 8);
      async_cp16(bbase + d, Bm + (size_t)(n0 + row) * K + kb + c8 * 8);
    }
  };

  v8f acc[4] = {};
  int nkb = K / KB;
  issue(0, 0);
  for (int kb = 0; kb < nkb; ++kb) {
    int cur = kb & 1;
    if (kb + 1 < nkb) { issue((kb + 1) * KB, cur ^ 1); ASYNC_WAIT(0x8); }
    else              { ASYNC_WAIT(0x0); }
    __syncthreads();
    const __bf16* arow = &Ash[cur][(wid * 16 + r) * LROW];
    const __bf16* brow = &Bsh[cur][r * LROW];
    #pragma unroll
    for (int ks = 0; ks < KB; ks += 32) {
      // batch all fragment loads, then fire 4 WMMAs back-to-back
      v16bf a = ldfragA(arow, ks, hs);
      v16bf bfr[4];
      #pragma unroll
      for (int t = 0; t < 4; ++t) bfr[t] = ldfragB(brow + t * 16 * LROW, ks, hs);
      #pragma unroll
      for (int t = 0; t < 4; ++t) acc[t] = wmma_bf16(a, bfr[t], acc[t]);
    }
    __syncthreads();   // all waves done reading before buffer is re-filled
  }

  #pragma unroll
  for (int t = 0; t < 4; ++t) {
    #pragma unroll
    for (int i = 0; i < 8; ++i) {
      int m = m0 + wid * 16 + i + 8 * hs;   // C/D layout: VGPR i, half-wave hs
      int n = n0 + t * 16 + r;
      float v = acc[t][i];
      if (bias) v += bias[n];
      size_t idx;
      if (chw) { int bb = m / HWPX, p = m - bb * HWPX;
                 idx = ((size_t)bb * N + n) * HWPX + p; }
      else       idx = (size_t)m * N + n;
      if (res)  v += res[idx];
      if (outF) outF[idx] = v;
      if (outB) outB[(size_t)m * N + n] = (__bf16)v;
    }
  }
}

// ---------------------------------------------------------------------------
// Fused GeGLU GEMM with async-to-LDS staging of A + both weight strips.
//   t[M,640] @ lin1_w -> (a|gate)[M,5120]; out = a*gelu_exact(gate), bf16
// ---------------------------------------------------------------------------
__global__ void gemm_geglu(const __bf16* __restrict__ A, const __bf16* __restrict__ Bm,
                           const float* __restrict__ bias, __bf16* __restrict__ out,
                           int M, int K) {
  __shared__ __bf16 Ash[2][64 * LROW];
  __shared__ __bf16 BAs[2][64 * LROW];
  __shared__ __bf16 BGs[2][64 * LROW];
  int tid = threadIdx.x;
  int lane = tid & 31, wid = tid >> 5;
  int m0 = blockIdx.x * 64;
  int n0 = blockIdx.y * 64;
  int r = lane & 15, hs = lane >> 4;

  unsigned ldsA0 = lds_byte_off(&Ash[0][0]);
  unsigned ldsBA = lds_byte_off(&BAs[0][0]);
  unsigned ldsBG = lds_byte_off(&BGs[0][0]);

  auto issue = [&](int kb, int buf) {
    unsigned ab = ldsA0 + (unsigned)buf * (64 * LROW * 2);
    unsigned b1 = ldsBA + (unsigned)buf * (64 * LROW * 2);
    unsigned b2 = ldsBG + (unsigned)buf * (64 * LROW * 2);
    #pragma unroll
    for (int j = 0; j < 4; ++j) {
      int e = j * 128 + tid;
      int row = e >> 3, c8 = e & 7;
      unsigned d = (unsigned)(row * LROW + c8 * 8) * 2;
      async_cp16(ab + d, A  + (size_t)(m0 + row) * K + kb + c8 * 8);
      async_cp16(b1 + d, Bm + (size_t)(n0 + row) * K + kb + c8 * 8);
      async_cp16(b2 + d, Bm + (size_t)(2560 + n0 + row) * K + kb + c8 * 8);
    }
  };

  v8f aa[4] = {}, gg[4] = {};
  int nkb = K / KB;
  issue(0, 0);
  for (int kb = 0; kb < nkb; ++kb) {
    int cur = kb & 1;
    if (kb + 1 < nkb) { issue((kb + 1) * KB, cur ^ 1); ASYNC_WAIT(0xC); }
    else              { ASYNC_WAIT(0x0); }
    __syncthreads();
    const __bf16* arow  = &Ash[cur][(wid * 16 + r) * LROW];
    const __bf16* b1row = &BAs[cur][r * LROW];
    const __bf16* b2row = &BGs[cur][r * LROW];
    #pragma unroll
    for (int ks = 0; ks < KB; ks += 32) {
      v16bf a = ldfragA(arow, ks, hs);
      v16bf f1[4], f2[4];
      #pragma unroll
      for (int t = 0; t < 4; ++t) {
        f1[t] = ldfragB(b1row + t * 16 * LROW, ks, hs);
        f2[t] = ldfragB(b2row + t * 16 * LROW, ks, hs);
      }
      #pragma unroll
      for (int t = 0; t < 4; ++t) {
        aa[t] = wmma_bf16(a, f1[t], aa[t]);
        gg[t] = wmma_bf16(a, f2[t], gg[t]);
      }
    }
    __syncthreads();
  }

  #pragma unroll
  for (int t = 0; t < 4; ++t) {
    #pragma unroll
    for (int i = 0; i < 8; ++i) {
      int m = m0 + wid * 16 + i + 8 * hs;
      int n = n0 + t * 16 + r;
      float va = aa[t][i] + bias[n];
      float vg = gg[t][i] + bias[2560 + n];
      float ge = 0.5f * vg * (1.0f + erff(vg * 0.70710678118654752f)); // exact gelu
      out[(size_t)m * 2560 + n] = (__bf16)(va * ge);
    }
  }
}

// ---------------------------------------------------------------------------
// Attention: one wave per (b, h, 16-query tile).
//  Q [b,h,1024,96] bf16, Kp [b,h,kv_rows,96] bf16, Vt [b,h,80,KVW] bf16.
//  Scores -> LDS (bf16, 16 x KVW), masked softmax (wave shfl reduce), P@V^T.
// ---------------------------------------------------------------------------
__global__ void attn_kernel(const __bf16* __restrict__ Q, const __bf16* __restrict__ Kp,
                            const __bf16* __restrict__ Vt, __bf16* __restrict__ out,
                            int kv_rows, int kv_len, int KVW, float scale) {
  extern __shared__ char smem[];
  __bf16* S = (__bf16*)smem;                      // [16][KVW]
  int lane = threadIdx.x;
  int r = lane & 15, hs = lane >> 4;
  int qt = blockIdx.x & 63;                       // 1024/16 query tiles
  int bh = blockIdx.x >> 6;
  int h = bh & 7, b = bh >> 3;

  // Phase 1: S = scale * Q K^T  (GEMM-K = DP = 96)
  const __bf16* qrow = Q + ((size_t)bh * HWPX + qt * 16 + r) * DP;
  int ktiles = kv_rows >> 4;
  for (int kt = 0; kt < ktiles; ++kt) {
    const __bf16* krow = Kp + ((size_t)bh * kv_rows + kt * 16 + r) * DP;
    v8f acc = {};
    v16bf af[3], bf[3];
    #pragma unroll
    for (int j = 0; j < 3; ++j) {
      af[j] = ldfragA(qrow, j * 32, hs);
      bf[j] = ldfragB(krow, j * 32, hs);
    }
    #pragma unroll
    for (int j = 0; j < 3; ++j) acc = wmma_bf16(af[j], bf[j], acc);
    #pragma unroll
    for (int i = 0; i < 8; ++i) {
      int ml = i + 8 * hs;
      S[ml * KVW + kt * 16 + r] = (__bf16)(acc[i] * scale);
    }
  }
  __syncthreads();

  // Phase 2: masked softmax.  Lane pair (r, r+16) shares row r, splits columns.
  int half = KVW >> 1;
  int c0 = hs * half;
  __bf16* srow = S + r * KVW;
  float mx = -1e30f;
  for (int c = 0; c < half; ++c) {
    int cg = c0 + c;
    if (cg < kv_len) { float v = (float)srow[cg]; mx = fmaxf(mx, v); }
  }
  mx = fmaxf(mx, __shfl_xor(mx, 16, 32));
  float sum = 0.f;
  for (int c = 0; c < half; ++c) {
    int cg = c0 + c;
    float e = (cg < kv_len) ? __expf((float)srow[cg] - mx) : 0.f;
    srow[cg] = (__bf16)e;
    sum += e;
  }
  sum += __shfl_xor(sum, 16, 32);
  float inv = 1.0f / sum;
  for (int c = 0; c < half; ++c) {
    int cg = c0 + c;
    srow[cg] = (__bf16)((float)srow[cg] * inv);
  }
  __syncthreads();

  // Phase 3: O[16,80] = P[16,KVW] @ Vt[80,KVW]^T — batch loads, burst WMMAs
  const __bf16* prow = S + r * KVW;
  v8f o[5] = {};
  for (int k = 0; k < KVW; k += 32) {
    v16bf a = ldfragA(prow, k, hs);
    v16bf vf[5];
    #pragma unroll
    for (int nt = 0; nt < 5; ++nt) {
      const __bf16* vrow = Vt + ((size_t)bh * DH + nt * 16 + r) * KVW;
      vf[nt] = ldfragB(vrow, k, hs);
    }
    #pragma unroll
    for (int nt = 0; nt < 5; ++nt) o[nt] = wmma_bf16(a, vf[nt], o[nt]);
  }
  #pragma unroll
  for (int nt = 0; nt < 5; ++nt) {
    #pragma unroll
    for (int i = 0; i < 8; ++i) {
      int q = qt * 16 + i + 8 * hs;
      int d = nt * 16 + r;
      out[((size_t)b * HWPX + q) * C_DIM + h * DH + d] = (__bf16)o[nt][i];
    }
  }
}

// ---------------------------------------------------------------------------
// Host orchestration
// ---------------------------------------------------------------------------
static inline unsigned cdiv(size_t a, size_t b) { return (unsigned)((a + b - 1) / b); }

extern "C" void kernel_launch(void* const* d_in, const int* in_sizes, int n_in,
                              void* d_out, int out_size, void* d_ws, size_t ws_size,
                              hipStream_t stream) {
  const float* x        = (const float*)d_in[0];
  const float* ctx      = (const float*)d_in[1];
  const float* gn_s     = (const float*)d_in[2];
  const float* gn_b     = (const float*)d_in[3];
  const float* conv1_w  = (const float*)d_in[4];
  const float* conv1_b  = (const float*)d_in[5];
  const float* ln1_s    = (const float*)d_in[6];
  const float* ln1_b    = (const float*)d_in[7];
  const float* sa_in_w  = (const float*)d_in[8];
  const float* sa_out_w = (const float*)d_in[9];
  const float* sa_out_b = (const float*)d_in[10];
  const float* ln2_s    = (const float*)d_in[11];
  const float* ln2_b    = (const float*)d_in[12];
  const float* ca_q_w   = (const float*)d_in[13];
  const float* ca_k_w   = (const float*)d_in[14];
  const float* ca_v_w   = (const float*)d_in[15];
  const float* ca_out_w = (const float*)d_in[16];
  const float* ca_out_b = (const float*)d_in[17];
  const float* ln3_s    = (const float*)d_in[18];
  const float* ln3_b    = (const float*)d_in[19];
  const float* lin1_w   = (const float*)d_in[20];
  const float* lin1_b   = (const float*)d_in[21];
  const float* lin2_w   = (const float*)d_in[22];
  const float* lin2_b   = (const float*)d_in[23];
  const float* co_w     = (const float*)d_in[24];
  const float* co_b     = (const float*)d_in[25];
  float* out = (float*)d_out;
  (void)in_sizes; (void)n_in; (void)out_size; (void)ws_size;

  size_t off = 0;
  auto alloc = [&](size_t bytes) -> void* {
    void* p = (char*)d_ws + off;
    off = (off + bytes + 255) & ~(size_t)255;
    return p;
  };

  __bf16* x0bf   = (__bf16*)alloc((size_t)MTOK * C_DIM * 2);
  __bf16* wc1    = (__bf16*)alloc((size_t)640 * 640 * 2);
  __bf16* wsain  = (__bf16*)alloc((size_t)1920 * 640 * 2);
  __bf16* wsaout = (__bf16*)alloc((size_t)640 * 640 * 2);
  __bf16* wcaq   = (__bf16*)alloc((size_t)640 * 640 * 2);
  __bf16* wcak   = (__bf16*)alloc((size_t)640 * 512 * 2);
  __bf16* wcav   = (__bf16*)alloc((size_t)640 * 512 * 2);
  __bf16* wcaout = (__bf16*)alloc((size_t)640 * 640 * 2);
  __bf16* wlin1  = (__bf16*)alloc((size_t)5120 * 640 * 2);
  __bf16* wlin2  = (__bf16*)alloc((size_t)640 * 2560 * 2);
  __bf16* wco    = (__bf16*)alloc((size_t)640 * 640 * 2);
  float*  x1     = (float*)alloc((size_t)MTOK * C_DIM * 4);
  float*  x2     = (float*)alloc((size_t)MTOK * C_DIM * 4);
  float*  x3     = (float*)alloc((size_t)MTOK * C_DIM * 4);
  __bf16* tbf    = (__bf16*)alloc((size_t)MTOK * C_DIM * 2);
  float*  qkv    = (float*)alloc((size_t)MTOK * 1920 * 4);   // reused for q2
  __bf16* Qp     = (__bf16*)alloc((size_t)64 * 1024 * DP * 2);
  __bf16* Kp     = (__bf16*)alloc((size_t)64 * 1024 * DP * 2);
  __bf16* Vt     = (__bf16*)alloc((size_t)64 * DH * 1024 * 2);
  __bf16* attnbf = (__bf16*)alloc((size_t)MTOK * C_DIM * 2);
  __bf16* ctxp   = (__bf16*)alloc((size_t)NB * 80 * 512 * 2);
  float*  kc     = (float*)alloc((size_t)640 * 640 * 4);
  float*  vc     = (float*)alloc((size_t)640 * 640 * 4);
  __bf16* Kp2    = (__bf16*)alloc((size_t)64 * 80 * DP * 2);
  __bf16* Vt2    = (__bf16*)alloc((size_t)64 * DH * 96 * 2);
  __bf16* ggbf   = (__bf16*)alloc((size_t)MTOK * 2560 * 2);
  __bf16* x4bf   = (__bf16*)alloc((size_t)MTOK * C_DIM * 2);

  const float scale = 0.11180339887498949f;  // 1/sqrt(80)

  // ---- weight prep (bf16, [N,K] layout) ----
  cast_kernel<<<cdiv(640 * 640, 256), 256, 0, stream>>>(conv1_w, wc1, (size_t)640 * 640);
  cast_kernel<<<cdiv(640 * 640, 256), 256, 0, stream>>>(co_w, wco, (size_t)640 * 640);
  wt_kernel<<<cdiv((size_t)640 * 1920, 256), 256, 0, stream>>>(sa_in_w, wsain, 640, 1920);
  wt_kernel<<<cdiv((size_t)640 * 640, 256), 256, 0, stream>>>(sa_out_w, wsaout, 640, 640);
  wt_kernel<<<cdiv((size_t)640 * 640, 256), 256, 0, stream>>>(ca_q_w, wcaq, 640, 640);
  wt_kernel<<<cdiv((size_t)512 * 640, 256), 256, 0, stream>>>(ca_k_w, wcak, 512, 640);
  wt_kernel<<<cdiv((size_t)512 * 640, 256), 256, 0, stream>>>(ca_v_w, wcav, 512, 640);
  wt_kernel<<<cdiv((size_t)640 * 640, 256), 256, 0, stream>>>(ca_out_w, wcaout, 640, 640);
  wt_kernel<<<cdiv((size_t)640 * 5120, 256), 256, 0, stream>>>(lin1_w, wlin1, 640, 5120);
  wt_kernel<<<cdiv((size_t)2560 * 640, 256), 256, 0, stream>>>(lin2_w, wlin2, 2560, 640);

  // ---- GroupNorm + conv1 ----
  gn_kernel<<<NB * 32, 256, 0, stream>>>(x, gn_s, gn_b, x0bf);
  gemm_nt<<<dim3(MTOK / 64, C_DIM / 64), 128, 0, stream>>>(
      x0bf, wc1, conv1_b, nullptr, x1, nullptr, MTOK, C_DIM, C_DIM, 0);

  // ---- self-attention ----
  ln_kernel<<<MTOK / 8, 256, 0, stream>>>(x1, ln1_s, ln1_b, tbf, MTOK);
  gemm_nt<<<dim3(MTOK / 64, 1920 / 64), 128, 0, stream>>>(
      tbf, wsain, nullptr, nullptr, qkv, nullptr, MTOK, 1920, C_DIM, 0);
  repack_qk_kernel<<<cdiv((size_t)64 * 1024 * DP, 256), 256, 0, stream>>>(qkv, Qp, 1024, 1920, 0);
  repack_qk_kernel<<<cdiv((size_t)64 * 1024 * DP, 256), 256, 0, stream>>>(qkv, Kp, 1024, 1920, 640);
  repack_vt_kernel<<<cdiv((size_t)64 * DH * 1024, 256), 256, 0, stream>>>(qkv, Vt, 1024, 1024, 1920, 1280);
  attn_kernel<<<64 * 64, 32, 16 * 1024 * 2, stream>>>(Qp, Kp, Vt, attnbf, 1024, 1024, 1024, scale);
  gemm_nt<<<dim3(MTOK / 64, C_DIM / 64), 128, 0, stream>>>(
      attnbf, wsaout, sa_out_b, x1, x2, nullptr, MTOK, C_DIM, C_DIM, 0);

  // ---- cross-attention ----
  ln_kernel<<<MTOK / 8, 256, 0, stream>>>(x2, ln2_s, ln2_b, tbf, MTOK);
  gemm_nt<<<dim3(MTOK / 64, C_DIM / 64), 128, 0, stream>>>(
      tbf, wcaq, nullptr, nullptr, qkv, nullptr, MTOK, C_DIM, C_DIM, 0);
  repack_qk_kernel<<<cdiv((size_t)64 * 1024 * DP, 256), 256, 0, stream>>>(qkv, Qp, 1024, 640, 0);
  ctxpad_kernel<<<cdiv((size_t)NB * 80 * 512, 256), 256, 0, stream>>>(ctx, ctxp);
  gemm_nt<<<dim3(640 / 64, C_DIM / 64), 128, 0, stream>>>(
      ctxp, wcak, nullptr, nullptr, kc, nullptr, 640, C_DIM, 512, 0);
  gemm_nt<<<dim3(640 / 64, C_DIM / 64), 128, 0, stream>>>(
      ctxp, wcav, nullptr, nullptr, vc, nullptr, 640, C_DIM, 512, 0);
  repack_qk_kernel<<<cdiv((size_t)64 * 80 * DP, 256), 256, 0, stream>>>(kc, Kp2, 80, 640, 0);
  repack_vt_kernel<<<cdiv((size_t)64 * DH * 96, 256), 256, 0, stream>>>(vc, Vt2, 80, 96, 640, 0);
  attn_kernel<<<64 * 64, 32, 16 * 96 * 2, stream>>>(Qp, Kp2, Vt2, attnbf, 80, 77, 96, scale);
  gemm_nt<<<dim3(MTOK / 64, C_DIM / 64), 128, 0, stream>>>(
      attnbf, wcaout, ca_out_b, x2, x3, nullptr, MTOK, C_DIM, C_DIM, 0);

  // ---- GeGLU FFN ----
  ln_kernel<<<MTOK / 8, 256, 0, stream>>>(x3, ln3_s, ln3_b, tbf, MTOK);
  gemm_geglu<<<dim3(MTOK / 64, 2560 / 64), 128, 0, stream>>>(tbf, wlin1, lin1_b, ggbf, MTOK, C_DIM);
  gemm_nt<<<dim3(MTOK / 64, C_DIM / 64), 128, 0, stream>>>(
      ggbf, wlin2, lin2_b, x3, nullptr, x4bf, MTOK, C_DIM, 2560, 0);

  // ---- final conv + long residual, stored in NCHW order ----
  gemm_nt<<<dim3(MTOK / 64, C_DIM / 64), 128, 0, stream>>>(
      x4bf, wco, co_b, x, out, nullptr, MTOK, C_DIM, C_DIM, 1);
}
